// STBlock_38963943309973
// MI455X (gfx1250) — matrix operand order repaired and checked by reading, not compiled, not run on hardware
//
#include <hip/hip_runtime.h>

// ---------- types ----------
typedef __bf16 bf16_t;
typedef __attribute__((ext_vector_type(16))) __bf16 v16bf;
typedef __attribute__((ext_vector_type(8)))  __bf16 v8bf;
typedef __attribute__((ext_vector_type(8)))  float  v8f;

union BFU { unsigned short u; bf16_t b; };

__device__ __forceinline__ bf16_t f2bf(float f) {
  unsigned u = __float_as_uint(f);
  u = (u + 0x7FFFu + ((u >> 16) & 1u)) >> 16;
  BFU t; t.u = (unsigned short)u; return t.b;
}
__device__ __forceinline__ float bf2f(bf16_t b) {
  BFU t; t.b = b; return __uint_as_float((unsigned)t.u << 16);
}
__device__ __forceinline__ v8f zero8() { v8f z = {0.f,0.f,0.f,0.f,0.f,0.f,0.f,0.f}; return z; }

__device__ __forceinline__ v8f wmma_bf16(v16bf a, v16bf b, v8f c) {
  return __builtin_amdgcn_wmma_f32_16x16x32_bf16(false, a, false, b, (short)0, c, false, false);
}

// A-operand fragment (16x32 bf16, row-major source, stride ld elements).
// lanes 0-15: rows 0-15, k = {0..7, 16..23}; lanes 16-31: k = {8..15, 24..31}.
__device__ __forceinline__ v16bf load_a_frag(const bf16_t* base, int ld) {
  const int lane = threadIdx.x & 31;
  const bf16_t* p = base + (lane & 15) * ld + ((lane & 16) ? 8 : 0);
  v8bf g0 = *(const v8bf*)p;
  v8bf g1 = *(const v8bf*)(p + 16);
  v16bf r;
#pragma unroll
  for (int i = 0; i < 8; ++i) { r[i] = g0[i]; r[i + 8] = g1[i]; }
  return r;
}
// B-operand fragment (32x16): source is [N][K] row-major (each output col n holds
// contiguous k). lanes 0-15: k = {0..7, 16..23}; lanes 16-31: k = {8..15, 24..31}.
__device__ __forceinline__ v16bf load_b_frag(const bf16_t* base, int ld) {
  const int lane = threadIdx.x & 31;
  const bf16_t* p = base + (lane & 15) * ld + ((lane & 16) ? 8 : 0);
  v8bf g0 = *(const v8bf*)p;
  v8bf g1 = *(const v8bf*)(p + 16);
  v16bf r;
#pragma unroll
  for (int i = 0; i < 8; ++i) { r[i] = g0[i]; r[i + 8] = g1[i]; }
  return r;
}

// ---------- generic WMMA GEMM: O[M,N] = epilogue(A[M,K] @ Wt[N,K]^T) ----------
template<int GELU, int HAS_RES, int ROWSCALE, int ACC_LOAD, int OUT32, int OUT16>
__global__ __launch_bounds__(256) void gemm_kernel(
    const bf16_t* __restrict__ A, const bf16_t* __restrict__ Wt,
    const float* __restrict__ bias, const float* __restrict__ residual,
    const float* __restrict__ rowscale, float* __restrict__ out32,
    bf16_t* __restrict__ out16, int M, int N, int K)
{
  __shared__ __align__(64) bf16_t sA[2][128 * 32];
  __shared__ __align__(64) bf16_t sB[2][128 * 32];
  const int tid = threadIdx.x;
  const int m0 = blockIdx.y * 128;
  const int n0 = blockIdx.x * 128;
  const int wave = tid >> 5;
  const int wm = (wave >> 2) * 64;   // 0 / 64
  const int wn = (wave & 3) * 32;    // 0 / 32 / 64 / 96

  v8f acc[4][2];
#pragma unroll
  for (int i = 0; i < 4; ++i)
#pragma unroll
    for (int j = 0; j < 2; ++j) acc[i][j] = zero8();

  const int KT = K >> 5;
  const int c0 = tid, c1 = tid + 256;
  const int row0 = c0 >> 2, ko0 = (c0 & 3) * 8;
  const int row1 = c1 >> 2, ko1 = (c1 & 3) * 8;
  const bf16_t* gA0 = A + (size_t)(m0 + row0) * K + ko0;
  const bf16_t* gA1 = A + (size_t)(m0 + row1) * K + ko1;
  const bf16_t* gB0 = Wt + (size_t)(n0 + row0) * K + ko0;
  const bf16_t* gB1 = Wt + (size_t)(n0 + row1) * K + ko1;

  *(v8bf*)&sA[0][row0 * 32 + ko0] = *(const v8bf*)(gA0);
  *(v8bf*)&sA[0][row1 * 32 + ko1] = *(const v8bf*)(gA1);
  *(v8bf*)&sB[0][row0 * 32 + ko0] = *(const v8bf*)(gB0);
  *(v8bf*)&sB[0][row1 * 32 + ko1] = *(const v8bf*)(gB1);
  __syncthreads();

  for (int kt = 0; kt < KT; ++kt) {
    if (kt + 1 < KT) {
      const int nb = (kt + 1) & 1;
      const size_t koff = (size_t)(kt + 1) * 32;
      *(v8bf*)&sA[nb][row0 * 32 + ko0] = *(const v8bf*)(gA0 + koff);
      *(v8bf*)&sA[nb][row1 * 32 + ko1] = *(const v8bf*)(gA1 + koff);
      *(v8bf*)&sB[nb][row0 * 32 + ko0] = *(const v8bf*)(gB0 + koff);
      *(v8bf*)&sB[nb][row1 * 32 + ko1] = *(const v8bf*)(gB1 + koff);
      if (kt + 2 < KT) {
        __builtin_prefetch(gA0 + koff + 32, 0, 1);   // global_prefetch_b8
        __builtin_prefetch(gB0 + koff + 32, 0, 1);
      }
    }
    const bf16_t* tA = sA[kt & 1];
    const bf16_t* tB = sB[kt & 1];
    v16bf b0 = load_b_frag(tB + (wn) * 32, 32);
    v16bf b1 = load_b_frag(tB + (wn + 16) * 32, 32);
#pragma unroll
    for (int i = 0; i < 4; ++i) {
      v16bf af = load_a_frag(tA + (wm + i * 16) * 32, 32);
      acc[i][0] = wmma_bf16(af, b0, acc[i][0]);
      acc[i][1] = wmma_bf16(af, b1, acc[i][1]);
    }
    __syncthreads();
  }

  const int lane = tid & 31;
  const int cn = lane & 15;
  const int rbase = (lane & 16) ? 8 : 0;
#pragma unroll
  for (int i = 0; i < 4; ++i) {
#pragma unroll
    for (int j = 0; j < 2; ++j) {
      const int gn = n0 + wn + j * 16 + cn;
      const float bv = bias ? bias[gn] : 0.0f;
#pragma unroll
      for (int r = 0; r < 8; ++r) {
        const int gm = m0 + wm + i * 16 + rbase + r;
        float v = acc[i][j][r] + bv;
        if (GELU) {
          float x3 = v * v * v;
          v = 0.5f * v * (1.0f + tanhf(0.7978845608f * (v + 0.044715f * x3)));
        }
        const size_t o = (size_t)gm * N + gn;
        if (HAS_RES) v += residual[o];
        if (ROWSCALE) {
          const float sc = rowscale[gm];
          v = ACC_LOAD ? (out32[o] + sc * v) : (sc * v);
        }
        if (OUT32) out32[o] = v;
        if (OUT16) out16[o] = f2bf(v);
      }
    }
  }
}

// ---------- spatial attention: one block per (bn, h); N=256, D=64 ----------
// Streaming online-softmax: 4 chunks of 64 key columns -> only 4 score tiles
// live at a time (keeps VGPRs < 256, avoids s_set_vgpr_msb pressure).
__global__ __launch_bounds__(256) void attn_spatial(
    const bf16_t* __restrict__ Qh, const bf16_t* __restrict__ Kh,
    const bf16_t* __restrict__ Vh, bf16_t* __restrict__ O)
{
  __shared__ __align__(64) bf16_t sVt[64 * 256];   // V transposed [d][seq]  (32KB)
  __shared__ __align__(64) bf16_t sP[8][16 * 64];  // per-wave P chunk       (16KB)
  const int h = blockIdx.x, bn = blockIdx.y;
  const size_t base = ((size_t)(bn * 16 + h)) * 256 * 64;
  const bf16_t* Qg = Qh + base;
  const bf16_t* Kg = Kh + base;
  const bf16_t* Vg = Vh + base;
  const int tid = threadIdx.x;
#pragma unroll 4
  for (int i = 0; i < 64; ++i) {
    int idx = tid + i * 256;
    int s = idx >> 6, d = idx & 63;
    sVt[d * 256 + s] = Vg[idx];
  }
  __syncthreads();
  const int wave = tid >> 5, lane = tid & 31;
  const int cn = lane & 15;
  const int rbase = (lane & 16) ? 8 : 0;
  bf16_t* myP = sP[wave];
  const float scale = 1.0f / 64.0f;
  for (int rb = 0; rb < 2; ++rb) {
    const int m0 = rb * 128 + wave * 16;
    v16bf qf0 = load_a_frag(Qg + (size_t)m0 * 64, 64);
    v16bf qf1 = load_a_frag(Qg + (size_t)m0 * 64 + 32, 64);
    float rm[8], rs[8];
#pragma unroll
    for (int r = 0; r < 8; ++r) { rm[r] = -1e30f; rs[r] = 0.f; }
    v8f oacc[4];
#pragma unroll
    for (int j = 0; j < 4; ++j) oacc[j] = zero8();

    for (int ch = 0; ch < 4; ++ch) {          // 64 key columns per chunk
      v8f sc[4];
#pragma unroll
      for (int u = 0; u < 4; ++u) {
        const int nt = ch * 4 + u;
        v8f a = zero8();
        a = wmma_bf16(qf0, load_b_frag(Kg + (size_t)nt * 16 * 64, 64), a);
        a = wmma_bf16(qf1, load_b_frag(Kg + (size_t)nt * 16 * 64 + 32, 64), a);
        sc[u] = a;
      }
      // online softmax update per row (rows live within a 16-lane half)
#pragma unroll
      for (int r = 0; r < 8; ++r) {
        float cm = -1e30f;
#pragma unroll
        for (int u = 0; u < 4; ++u) cm = fmaxf(cm, sc[u][r]);
#pragma unroll
        for (int dd = 1; dd < 16; dd <<= 1) cm = fmaxf(cm, __shfl_xor(cm, dd, 32));
        const float mnew = fmaxf(rm[r], cm * scale);
        const float alpha = __expf(rm[r] - mnew);
        rm[r] = mnew;
        float csum = 0.f;
#pragma unroll
        for (int u = 0; u < 4; ++u) {
          float e = __expf(sc[u][r] * scale - mnew);
          sc[u][r] = e;
          csum += e;
        }
#pragma unroll
        for (int dd = 1; dd < 16; dd <<= 1) csum += __shfl_xor(csum, dd, 32);
        rs[r] = rs[r] * alpha + csum;
#pragma unroll
        for (int j = 0; j < 4; ++j) oacc[j][r] *= alpha;
      }
      // stage P chunk (C-layout -> A-layout via per-wave LDS)
#pragma unroll
      for (int u = 0; u < 4; ++u)
#pragma unroll
        for (int r = 0; r < 8; ++r)
          myP[(rbase + r) * 64 + u * 16 + cn] = f2bf(sc[u][r]);
      asm volatile("s_wait_dscnt 0" ::: "memory");
#pragma unroll
      for (int kc = 0; kc < 2; ++kc) {
        v16bf pf = load_a_frag(myP + kc * 32, 64);
#pragma unroll
        for (int j = 0; j < 4; ++j)
          oacc[j] = wmma_bf16(pf,
              load_b_frag(sVt + j * 16 * 256 + ch * 64 + kc * 32, 256), oacc[j]);
      }
      asm volatile("" ::: "memory");
    }
    // finalize: divide by softmax denominator, store
#pragma unroll
    for (int j = 0; j < 4; ++j)
#pragma unroll
      for (int r = 0; r < 8; ++r) {
        const size_t tok = (size_t)bn * 256 + m0 + rbase + r;
        O[tok * 1024 + h * 64 + j * 16 + cn] = f2bf(oacc[j][r] / rs[r]);
      }
  }
}

// ---------- temporal causal attention: one wave per (bs, h); T=16, D=64 ----------
__global__ __launch_bounds__(256) void attn_temporal(
    const bf16_t* __restrict__ Qh, const bf16_t* __restrict__ Kh,
    const bf16_t* __restrict__ Vh, bf16_t* __restrict__ O)
{
  __shared__ __align__(64) bf16_t sVt[8][64 * 32];
  __shared__ __align__(64) bf16_t sP[8][16 * 32];
  const int tid = threadIdx.x, wave = tid >> 5, lane = tid & 31;
  const int pair = blockIdx.x * 8 + wave;
  const int bs = pair >> 4, h = pair & 15;
  const size_t base = ((size_t)(bs * 16 + h)) * 16 * 64;
  const bf16_t* Qg = Qh + base;
  const bf16_t* Kg = Kh + base;
  const bf16_t* Vg = Vh + base;
  bf16_t* myVt = sVt[wave];
  bf16_t* myP = sP[wave];
  const bf16_t zb = f2bf(0.0f);
#pragma unroll
  for (int i = 0; i < 64; ++i) myVt[lane + i * 32] = zb;   // zero incl. K-pad
#pragma unroll
  for (int i = 0; i < 16; ++i) myP[lane + i * 32] = zb;    // zero incl. K-pad
#pragma unroll
  for (int i = 0; i < 32; ++i) {
    int idx = lane * 32 + i;
    int t = idx >> 6, d = idx & 63;
    myVt[d * 32 + t] = Vg[idx];
  }
  const int cn = lane & 15, rbase = (lane & 16) ? 8 : 0;
  v8f s8 = zero8();
  s8 = wmma_bf16(load_a_frag(Qg, 64), load_b_frag(Kg, 64), s8);
  s8 = wmma_bf16(load_a_frag(Qg + 32, 64), load_b_frag(Kg + 32, 64), s8);
  const float scale = 1.0f / 64.0f;
  float w[8];
#pragma unroll
  for (int r = 0; r < 8; ++r) {
    const int rr = rbase + r;
    float p = (cn <= rr) ? s8[r] * scale : -1e30f;   // causal mask
    float m = p;
#pragma unroll
    for (int dd = 1; dd < 16; dd <<= 1) m = fmaxf(m, __shfl_xor(m, dd, 32));
    float e = __expf(p - m);
    float ssum = e;
#pragma unroll
    for (int dd = 1; dd < 16; dd <<= 1) ssum += __shfl_xor(ssum, dd, 32);
    w[r] = e / ssum;
  }
#pragma unroll
  for (int r = 0; r < 8; ++r) myP[(rbase + r) * 32 + cn] = f2bf(w[r]);
  asm volatile("s_wait_dscnt 0" ::: "memory");
  v16bf pf = load_a_frag(myP, 32);
  v8f oacc[4];
#pragma unroll
  for (int j = 0; j < 4; ++j)
    oacc[j] = wmma_bf16(pf, load_b_frag(myVt + j * 16 * 32, 32), zero8());
#pragma unroll
  for (int j = 0; j < 4; ++j)
#pragma unroll
    for (int r = 0; r < 8; ++r) {
      const size_t tok = (size_t)bs * 16 + rbase + r;
      O[tok * 1024 + h * 64 + j * 16 + cn] = f2bf(oacc[j][r]);
    }
}

// ---------- qk-LayerNorm + head-major split of packed QKV ----------
__global__ void qkv_prep(const bf16_t* __restrict__ qkv, const float* __restrict__ qg,
                         const float* __restrict__ qb, const float* __restrict__ kg,
                         const float* __restrict__ kb, bf16_t* __restrict__ Qh,
                         bf16_t* __restrict__ Kh, bf16_t* __restrict__ Vh, int L)
{
  const int gid = blockIdx.x * 256 + threadIdx.x;   // token*16 + head
  const int token = gid >> 4, h = gid & 15;
  const int bn = token / L, pos = token - bn * L;
  const size_t inb = (size_t)token * 3072 + h * 64;
  const size_t outb = (((size_t)bn * 16 + h) * L + pos) * 64;
  {
    float s = 0.f, s2 = 0.f;
    for (int d = 0; d < 64; ++d) { float v = bf2f(qkv[inb + d]); s += v; s2 += v * v; }
    const float mean = s * (1.f / 64.f);
    const float inv = rsqrtf(s2 * (1.f / 64.f) - mean * mean + 1e-5f);
    for (int d = 0; d < 64; ++d) {
      float v = bf2f(qkv[inb + d]);
      Qh[outb + d] = f2bf((v - mean) * inv * qg[d] + qb[d]);
    }
  }
  {
    float s = 0.f, s2 = 0.f;
    for (int d = 0; d < 64; ++d) { float v = bf2f(qkv[inb + 1024 + d]); s += v; s2 += v * v; }
    const float mean = s * (1.f / 64.f);
    const float inv = rsqrtf(s2 * (1.f / 64.f) - mean * mean + 1e-5f);
    for (int d = 0; d < 64; ++d) {
      float v = bf2f(qkv[inb + 1024 + d]);
      Kh[outb + d] = f2bf((v - mean) * inv * kg[d] + kb[d]);
    }
  }
  for (int d = 0; d < 64; ++d) Vh[outb + d] = qkv[inb + 2048 + d];
}

// ---------- MoE gating ----------
__global__ void gate_init(float* a) { if (blockIdx.x == 0 && threadIdx.x < 8) a[threadIdx.x] = 0.f; }

__global__ void gate_kernel(const float* __restrict__ X, const float* __restrict__ gw,
                            float* __restrict__ wts, float* auxbuf, int NT)
{
  const int t = blockIdx.x * 256 + threadIdx.x;
  if (t >= NT) return;
  const float* xr = X + (size_t)t * 1024;
  float l[4] = {0.f, 0.f, 0.f, 0.f};
  for (int k = 0; k < 1024; ++k) {
    const float xv = xr[k];
    const float* g = gw + (size_t)k * 4;
    l[0] += xv * g[0]; l[1] += xv * g[1]; l[2] += xv * g[2]; l[3] += xv * g[3];
  }
  const float lm = fmaxf(fmaxf(l[0], l[1]), fmaxf(l[2], l[3]));
  float p[4]; float sum = 0.f;
  for (int e = 0; e < 4; ++e) { p[e] = __expf(l[e] - lm); sum += p[e]; }
  for (int e = 0; e < 4; ++e) p[e] /= sum;
  int i0 = 0;
  for (int e = 1; e < 4; ++e) if (p[e] > p[i0]) i0 = e;
  int i1 = (i0 == 0) ? 1 : 0;
  for (int e = 0; e < 4; ++e) if (e != i0 && p[e] > p[i1]) i1 = e;
  const float wsum = p[i0] + p[i1];
  for (int e = 0; e < 4; ++e) {
    const float w = (e == i0) ? p[i0] / wsum : ((e == i1) ? p[i1] / wsum : 0.f);
    wts[(size_t)e * NT + t] = w;
  }
  atomicAdd(&auxbuf[i0], 1.0f);
  atomicAdd(&auxbuf[i1], 1.0f);
  for (int e = 0; e < 4; ++e) atomicAdd(&auxbuf[4 + e], p[e]);
}

__global__ void aux_fin(const float* auxb, float* dst, int NT)
{
  if (blockIdx.x == 0 && threadIdx.x == 0) {
    float a = 0.f;
    const float invN = 1.0f / (float)NT;
    for (int e = 0; e < 4; ++e) a += (auxb[e] * invN) * (auxb[4 + e] * invN);
    dst[0] = 4.0f * a;
  }
}

// ---------- layout / dtype helpers ----------
__global__ void conv_bf16(const float* __restrict__ in, bf16_t* __restrict__ out, size_t n)
{
  const size_t i = (size_t)blockIdx.x * 256 + threadIdx.x;
  if (i < n) out[i] = f2bf(in[i]);
}
__global__ void conv_bf16_T(const float* __restrict__ in, bf16_t* __restrict__ out, int R, int Cc)
{
  const size_t i = (size_t)blockIdx.x * 256 + threadIdx.x;
  if (i < (size_t)R * Cc) {
    const int cc = (int)(i % Cc);
    const int rr = (int)(i / Cc);
    out[(size_t)cc * R + rr] = f2bf(in[i]);
  }
}
// [B,T,S,C] -> [B,S,T,C] with bf16 mirror
__global__ void transpose_bt(const float* __restrict__ in, float* __restrict__ out,
                             bf16_t* __restrict__ out16)
{
  const size_t idx = (size_t)blockIdx.x * 256 + threadIdx.x;
  const int c = (int)(idx & 1023);
  const size_t r = idx >> 10;
  const int s = (int)(r & 255);
  const size_t bt = r >> 8;
  const int t = (int)(bt & 15);
  const int b = (int)(bt >> 4);
  const size_t oi = ((((size_t)b * 256 + s) * 16 + t) * 1024) + c;
  const float v = in[idx];
  out[oi] = v;
  out16[oi] = f2bf(v);
}
// out[B,T,S,C] = (xt + moe)[B,S,T,C]
__global__ void final_out(const float* __restrict__ xt, const float* __restrict__ moe,
                          float* __restrict__ out)
{
  const size_t idx = (size_t)blockIdx.x * 256 + threadIdx.x;
  const int c = (int)(idx & 1023);
  const size_t r = idx >> 10;
  const int s = (int)(r & 255);
  const size_t bt = r >> 8;
  const int t = (int)(bt & 15);
  const int b = (int)(bt >> 4);
  const size_t si = ((((size_t)b * 256 + s) * 16 + t) * 1024) + c;
  out[idx] = xt[si] + moe[si];
}

// ---------- orchestration ----------
extern "C" void kernel_launch(void* const* d_in, const int* in_sizes, int n_in,
                              void* d_out, int out_size, void* d_ws, size_t ws_size,
                              hipStream_t stream)
{
  (void)in_sizes; (void)n_in; (void)out_size; (void)ws_size;
  const float* x         = (const float*)d_in[0];
  const float* sa_qkv_w  = (const float*)d_in[1];
  const float* sa_proj_w = (const float*)d_in[2];
  const float* sa_proj_b = (const float*)d_in[3];
  const float* sa_qg = (const float*)d_in[4];
  const float* sa_qb = (const float*)d_in[5];
  const float* sa_kg = (const float*)d_in[6];
  const float* sa_kb = (const float*)d_in[7];
  const float* ta_qkv_w  = (const float*)d_in[8];
  const float* ta_proj_w = (const float*)d_in[9];
  const float* ta_proj_b = (const float*)d_in[10];
  const float* ta_qg = (const float*)d_in[11];
  const float* ta_qb = (const float*)d_in[12];
  const float* ta_kg = (const float*)d_in[13];
  const float* ta_kb = (const float*)d_in[14];
  const float* gate_w = (const float*)d_in[15];
  const float* w1 = (const float*)d_in[16];
  const float* b1 = (const float*)d_in[17];
  const float* w2 = (const float*)d_in[18];
  const float* b2 = (const float*)d_in[19];
  float* out = (float*)d_out;

  const int NT = 16384;          // B*T*S tokens
  const int C = 1024;

  char* ws = (char*)d_ws;
  size_t off = 0;
  auto alloc = [&](size_t bytes) -> char* {
    char* p = ws + off;
    off += (bytes + 255) & ~(size_t)255;
    return p;
  };
  bf16_t* wqkv_sa = (bf16_t*)alloc((size_t)3072 * 1024 * 2);
  bf16_t* wproj_sa = (bf16_t*)alloc((size_t)1024 * 1024 * 2);
  bf16_t* wqkv_ta = (bf16_t*)alloc((size_t)3072 * 1024 * 2);
  bf16_t* wproj_ta = (bf16_t*)alloc((size_t)1024 * 1024 * 2);
  bf16_t* ww1 = (bf16_t*)alloc((size_t)4 * 1024 * 1024 * 2);
  bf16_t* ww2 = (bf16_t*)alloc((size_t)4 * 1024 * 1024 * 2);
  bf16_t* XB  = (bf16_t*)alloc((size_t)NT * C * 2);
  bf16_t* QKVB = (bf16_t*)alloc((size_t)NT * 3 * C * 2);
  bf16_t* QH = (bf16_t*)alloc((size_t)NT * C * 2);
  bf16_t* KH = (bf16_t*)alloc((size_t)NT * C * 2);
  bf16_t* VH = (bf16_t*)alloc((size_t)NT * C * 2);
  bf16_t* AO = (bf16_t*)alloc((size_t)NT * C * 2);
  float* XSC = (float*)alloc((size_t)NT * C * 4);
  float* XTC = (float*)alloc((size_t)NT * C * 4);
  bf16_t* HB = (bf16_t*)alloc((size_t)NT * C * 2);
  float* MOE = (float*)alloc((size_t)NT * C * 4);
  float* WTS = (float*)alloc((size_t)4 * NT * 4);
  float* AUXB = (float*)alloc(64);

  dim3 blk(256);

  // ----- weight conversion (bf16, Wt = [N][K] layout) -----
  conv_bf16<<<3072 * 1024 / 256, blk, 0, stream>>>(sa_qkv_w, wqkv_sa, (size_t)3072 * 1024);
  conv_bf16<<<1024 * 1024 / 256, blk, 0, stream>>>(sa_proj_w, wproj_sa, (size_t)1024 * 1024);
  conv_bf16<<<3072 * 1024 / 256, blk, 0, stream>>>(ta_qkv_w, wqkv_ta, (size_t)3072 * 1024);
  conv_bf16<<<1024 * 1024 / 256, blk, 0, stream>>>(ta_proj_w, wproj_ta, (size_t)1024 * 1024);
  for (int e = 0; e < 4; ++e) {
    conv_bf16_T<<<1024 * 1024 / 256, blk, 0, stream>>>(
        w1 + (size_t)e * 1024 * 1024, ww1 + (size_t)e * 1024 * 1024, 1024, 1024);
    conv_bf16_T<<<1024 * 1024 / 256, blk, 0, stream>>>(
        w2 + (size_t)e * 1024 * 1024, ww2 + (size_t)e * 1024 * 1024, 1024, 1024);
  }
  conv_bf16<<<NT * C / 256, blk, 0, stream>>>(x, XB, (size_t)NT * C);

  // ----- spatial attention -----
  gemm_kernel<0,0,0,0,0,1><<<dim3(3072 / 128, NT / 128), blk, 0, stream>>>(
      XB, wqkv_sa, nullptr, nullptr, nullptr, nullptr, QKVB, NT, 3072, 1024);
  qkv_prep<<<NT * 16 / 256, blk, 0, stream>>>(QKVB, sa_qg, sa_qb, sa_kg, sa_kb, QH, KH, VH, 256);
  attn_spatial<<<dim3(16, 64), blk, 0, stream>>>(QH, KH, VH, AO);
  gemm_kernel<0,1,0,0,1,0><<<dim3(1024 / 128, NT / 128), blk, 0, stream>>>(
      AO, wproj_sa, sa_proj_b, x, nullptr, XSC, nullptr, NT, 1024, 1024);

  // ----- reorder to [B,S,T,C] for temporal pass -----
  transpose_bt<<<NT * C / 256, blk, 0, stream>>>(XSC, XTC, XB);

  // ----- temporal causal attention -----
  gemm_kernel<0,0,0,0,0,1><<<dim3(3072 / 128, NT / 128), blk, 0, stream>>>(
      XB, wqkv_ta, nullptr, nullptr, nullptr, nullptr, QKVB, NT, 3072, 1024);
  qkv_prep<<<NT * 16 / 256, blk, 0, stream>>>(QKVB, ta_qg, ta_qb, ta_kg, ta_kb, QH, KH, VH, 16);
  attn_temporal<<<2048, blk, 0, stream>>>(QH, KH, VH, AO);
  gemm_kernel<0,1,0,0,1,1><<<dim3(1024 / 128, NT / 128), blk, 0, stream>>>(
      AO, wproj_ta, ta_proj_b, XTC, nullptr, XSC, XB, NT, 1024, 1024);

  // ----- MoE gating + aux loss -----
  gate_init<<<1, 32, 0, stream>>>(AUXB);
  gate_kernel<<<NT / 256, blk, 0, stream>>>(XSC, gate_w, WTS, AUXB, NT);
  aux_fin<<<1, 32, 0, stream>>>(AUXB, out + (size_t)NT * C, NT);

  // ----- MoE experts (dense, gate-weighted accumulate) -----
  for (int e = 0; e < 4; ++e) {
    gemm_kernel<1,0,0,0,0,1><<<dim3(1024 / 128, NT / 128), blk, 0, stream>>>(
        XB, ww1 + (size_t)e * 1024 * 1024, b1 + (size_t)e * 1024,
        nullptr, nullptr, nullptr, HB, NT, 1024, 1024);
    if (e == 0) {
      gemm_kernel<0,0,1,0,1,0><<<dim3(1024 / 128, NT / 128), blk, 0, stream>>>(
          HB, ww2 + (size_t)e * 1024 * 1024, b2 + (size_t)e * 1024,
          nullptr, WTS + (size_t)e * NT, MOE, nullptr, NT, 1024, 1024);
    } else {
      gemm_kernel<0,0,1,1,1,0><<<dim3(1024 / 128, NT / 128), blk, 0, stream>>>(
          HB, ww2 + (size_t)e * 1024 * 1024, b2 + (size_t)e * 1024,
          nullptr, WTS + (size_t)e * NT, MOE, nullptr, NT, 1024, 1024);
    }
  }

  // ----- residual + back-transpose to [B,T,S,C] -----
  final_out<<<NT * C / 256, blk, 0, stream>>>(XSC, MOE, out);
}
